// Shared_Layer_82214263980115
// MI455X (gfx1250) — compile-verified, matrix operand order checked by reading, not run on hardware
//
#include <hip/hip_runtime.h>
#include <hip/hip_bf16.h>
#include <math.h>

typedef __attribute__((ext_vector_type(16))) __bf16 v16bf;
typedef __attribute__((ext_vector_type(8)))  __bf16 v8bf;
typedef __attribute__((ext_vector_type(8)))  float  v8f;
typedef __attribute__((ext_vector_type(4)))  unsigned int u32x4;
typedef __attribute__((ext_vector_type(8)))  int i32x8;
typedef __attribute__((ext_vector_type(4)))  int i32x4;

#define DEV __device__ __forceinline__
#define CUTOFF_F 5.0f
#define PI_F 3.14159265358979f

// LDS layout for the persistent edge kernel (dynamic shared memory).
#define WC_PITCH   40                      // 32 + 8 pad (bf16 elements)
#define W_PITCH    136                     // 128 + 8 pad (bf16 elements)
#define SM_WC_B    (128 * WC_PITCH * 2)    // 10240
#define SM_WE2_B   (128 * W_PITCH * 2)     // 34816
#define SM_WC1_B   (64 * W_PITCH * 2)      // 17408
#define SM_HB_OFF  (SM_WC_B + SM_WE2_B + SM_WC1_B)          // 62464
#define SM_TOTAL   (SM_HB_OFF + 4 * (16 * 132 + 16) * 4)    // 96512 bytes

DEV float siluf(float v) { return v * (1.0f / (1.0f + __expf(-v))); }

DEV v8f zero8() {
  v8f z;
#pragma unroll
  for (int i = 0; i < 8; ++i) z[i] = 0.0f;
  return z;
}

// Load a 16-element bf16 A/B fragment for wmma 16x16x32:
// elements [kb..kb+7] and [kb+16..kb+23] of a K-contiguous row. base = row + kb (16B aligned).
DEV v16bf ldv16g(const __bf16* base) {
  const v8bf* p = (const v8bf*)base;
  v8bf lo = p[0];
  v8bf hi = p[2];
  v16bf r;
#pragma unroll
  for (int i = 0; i < 8; ++i) { r[i] = lo[i]; r[8 + i] = hi[i]; }
  return r;
}

// Same fragment, sourced from an f32 row in LDS with on-the-fly bf16 convert.
DEV v16bf ldv16_lds_f32(const float* rowp, int kb) {
  v16bf r;
#pragma unroll
  for (int i = 0; i < 8; ++i) {
    r[i]     = (__bf16)rowp[kb + i];
    r[8 + i] = (__bf16)rowp[kb + 16 + i];
  }
  return r;
}

DEV v8f wmma_bf16(v16bf a, v16bf b, v8f c) {
  return __builtin_amdgcn_wmma_f32_16x16x32_bf16(false, a, false, b, (short)0, c, false, false);
}

DEV float leg_score(float ang) {
  const float c  = __cosf(ang);
  const float p2 = (3.0f * c * c - 1.0f) * 0.5f;
  const float p3 = (5.0f * c * p2 - 2.0f * c) * (1.0f / 3.0f);
  return 0.25f * (1.0f + fabsf(c) + fabsf(p2) + fabsf(p3));
}

// Tensor Data Mover: DMA nelem bf16 elements global->LDS as a 1-row tensor,
// inserting pad_amount(+1) DWORDs of LDS padding every 2^(pad_interval+1) DWORDs
// (hardware row padding for bank-conflict-free B-fragment reads).
// D# per CDNA5 ISA (group0 128b, group1 256b).
// This toolchain (clang-23 / therock-10.0 headers) exposes the 6-arg builtin:
// (u32x4 g0, i32x8 g1, i32x4 g2, i32x4 g3, i32x8 extra, i32 cpol).
DEV void tdm_load_1d_padded(const __bf16* gsrc, const __bf16* lds_dst,
                            unsigned nelem, unsigned pad_interval_code,
                            unsigned pad_amount_code) {
  const unsigned long long ga = (unsigned long long)(uintptr_t)gsrc;
  const unsigned lds = (unsigned)(uintptr_t)lds_dst;  // addr[31:0] == LDS byte offset
  u32x4 g0;
  g0[0] = 1u;                                          // count=1 valid descriptor
  g0[1] = lds;                                         // lds_addr
  g0[2] = (unsigned)(ga & 0xFFFFFFFFu);                // global_addr[31:0]
  g0[3] = (unsigned)((ga >> 32) & 0x1FFFFFFu) | (2u << 30);  // ga[56:32] | type=2
  i32x8 g1;
  g1[0] = (int)((1u << 16) | (1u << 20) |              // data_size=2B, pad_enable
                (pad_interval_code << 22) | (pad_amount_code << 25));
  g1[1] = (int)((nelem & 0xFFFFu) << 16);              // tensor_dim0[15:0] @ bits 63:48
  g1[2] = (int)(((nelem >> 16) & 0xFFFFu) | (1u << 16)); // tensor_dim0 hi | tensor_dim1=1
  g1[3] = (int)((nelem & 0xFFFFu) << 16);              // tile_dim0 @ bits 127:112
  g1[4] = 1;                                           // tile_dim1=1, tile_dim2=0
  g1[5] = (int)nelem;                                  // tensor_dim0_stride[31:0]
  g1[6] = 0;
  g1[7] = 0;
  i32x4 z4;
  z4[0] = 0; z4[1] = 0; z4[2] = 0; z4[3] = 0;
  i32x8 z8;
#pragma unroll
  for (int i = 0; i < 8; ++i) z8[i] = 0;
  __builtin_amdgcn_tensor_load_to_lds(g0, g1, z4, z4, z8, 0);
}

// ---------------------------------------------------------------- utilities

__global__ void k_zero(float* __restrict__ p, long n) {
  long i = (long)blockIdx.x * blockDim.x + threadIdx.x;
  if (i < n) p[i] = 0.0f;
}

__global__ void k_cvt_bf16(const float* __restrict__ src, __bf16* __restrict__ dst, long n) {
  long i = (long)blockIdx.x * blockDim.x + threadIdx.x;
  if (i < n) dst[i] = (__bf16)src[i];
}

// Transpose + convert all weight matrices into n-major / K-contiguous bf16 tables.
__global__ void k_prep_w(const float* __restrict__ We1, const float* __restrict__ We2,
                         const float* __restrict__ Wc1, const float* __restrict__ Wn,
                         __bf16* __restrict__ WaT, __bf16* __restrict__ WbT,
                         __bf16* __restrict__ WcT, __bf16* __restrict__ We2T,
                         __bf16* __restrict__ Wc1T, __bf16* __restrict__ WnaT,
                         __bf16* __restrict__ WnbT) {
  int i = blockIdx.x * blockDim.x + threadIdx.x;
  if (i < 5 * 16384) {                       // five 128x128 tables
    int w = i / 16384, r = i % 16384;
    int n = r / 128, k = r % 128;
    float v;
    __bf16* d;
    if (w == 0)      { v = We1[k * 128 + n];         d = WaT;  }
    else if (w == 1) { v = We1[(128 + k) * 128 + n]; d = WbT;  }
    else if (w == 2) { v = We2[k * 128 + n];         d = We2T; }
    else if (w == 3) { v = Wn[k * 128 + n];          d = WnaT; }
    else             { v = Wn[(128 + k) * 128 + n];  d = WnbT; }
    d[n * 128 + k] = (__bf16)v;
  } else if (i < 5 * 16384 + 128 * 32) {     // extras weights, K padded 19 -> 32
    int r = i - 5 * 16384;
    int n = r / 32, k = r % 32;
    float v = (k < 19) ? We1[(256 + k) * 128 + n] : 0.0f;
    WcT[n * 32 + k] = (__bf16)v;
  } else if (i < 5 * 16384 + 128 * 32 + 64 * 128) {
    int r = i - (5 * 16384 + 128 * 32);
    int n = r / 128, k = r % 128;
    Wc1T[n * 128 + k] = (__bf16)Wc1[k * 128 + n];
  }
}

// ------------------------------------------------- node-side pre-GEMM: P, Q

__global__ __launch_bounds__(256) void k_nodePQ(
    const __bf16* __restrict__ xbf, const __bf16* __restrict__ WaT,
    const __bf16* __restrict__ WbT, const float* __restrict__ be1,
    float* __restrict__ P, float* __restrict__ Q, int N) {
  const int lane  = threadIdx.x & 31;
  const int wave  = threadIdx.x >> 5;
  const int tile  = blockIdx.x * 8 + wave;
  const int row0  = tile * 16;
  if (row0 >= N) return;
  const int nlo   = lane & 15;
  const int khalf = (lane >> 4) << 3;   // 0 or 8
  int arow = row0 + nlo; if (arow >= N) arow = N - 1;
  const __bf16* xr = xbf + (long)arow * 128;

  v8f accP[8], accQ[8];
#pragma unroll
  for (int nt = 0; nt < 8; ++nt) { accP[nt] = zero8(); accQ[nt] = zero8(); }

#pragma unroll
  for (int kc = 0; kc < 4; ++kc) {
    const int kb = kc * 32 + khalf;
    v16bf a = ldv16g(xr + kb);
#pragma unroll
    for (int nt = 0; nt < 8; ++nt) {
      const int n = nt * 16 + nlo;
      accP[nt] = wmma_bf16(a, ldv16g(WaT + n * 128 + kb), accP[nt]);
      accQ[nt] = wmma_bf16(a, ldv16g(WbT + n * 128 + kb), accQ[nt]);
    }
  }
  const int mbase = (lane < 16) ? 0 : 8;
#pragma unroll
  for (int nt = 0; nt < 8; ++nt) {
    const int n  = nt * 16 + nlo;
    const float b1 = be1[n];
#pragma unroll
    for (int j = 0; j < 8; ++j) {
      const int r = row0 + mbase + j;
      if (r < N) {
        P[(long)r * 128 + n] = accP[nt][j] + b1;
        Q[(long)r * 128 + n] = accQ[nt][j];
      }
    }
  }
}

// ------------------------------------------------------------- edge kernel
// Persistent: TDM-stages the three edge weight tables into LDS once, then
// grid-strides over 16-edge tiles.

__global__ __launch_bounds__(128) void k_edge(
    const float* __restrict__ pos, const float* __restrict__ charge,
    const float* __restrict__ edge_attr, const int* __restrict__ edge_index,
    const float* __restrict__ P, const float* __restrict__ Q,
    const float* __restrict__ g1, const float* __restrict__ bt1,
    const __bf16* __restrict__ WcT, const __bf16* __restrict__ We2T,
    const float* __restrict__ be2, const __bf16* __restrict__ Wc1T,
    const float* __restrict__ bc1, const float* __restrict__ Wc2,
    const float* __restrict__ bc2, float* __restrict__ agg,
    float* __restrict__ cnt, float* __restrict__ delta, long E) {
  extern __shared__ char smem[];
  __bf16* sWc  = (__bf16*)smem;                            // 128 x WC_PITCH
  __bf16* sWe2 = (__bf16*)(smem + SM_WC_B);                // 128 x W_PITCH
  __bf16* sWc1 = (__bf16*)(smem + SM_WC_B + SM_WE2_B);     // 64  x W_PITCH
  float*  hall = (float*)(smem + SM_HB_OFF);

  const int lane = threadIdx.x & 31;
  const int wave = threadIdx.x >> 5;
  float* hb = hall + wave * (16 * 132 + 16);
  float* cb = hb + 16 * 132;

  // Stage weights LDS-side via the Tensor Data Mover (wave 0 issues; all wait).
  if (wave == 0) {
    tdm_load_1d_padded(WcT,  sWc,  128 * 32,  3, 3);  // rows of 16 DW + 4 DW pad
    tdm_load_1d_padded(We2T, sWe2, 128 * 128, 5, 3);  // rows of 64 DW + 4 DW pad
    tdm_load_1d_padded(Wc1T, sWc1, 64 * 128,  5, 3);
    __builtin_amdgcn_s_wait_tensorcnt(0);
  }
  __syncthreads();

  const int  nlo   = lane & 15;
  const int  khalf = (lane >> 4) << 3;
  const int  mbase = (lane < 16) ? 0 : 8;
  const int  chalf = (lane < 16) ? 0 : 64;
  const long ntile = (E + 15) >> 4;
  const long tstep = (long)gridDim.x * 4;

  for (long tile = (long)blockIdx.x * 4 + wave; tile < ntile; tile += tstep) {
    const long e  = tile * 16 + nlo;
    const bool ev = (e < E);
    const long ec = ev ? e : (E - 1);

    const int src = edge_index[ec];
    const int dst = edge_index[E + ec];
    const float rx = pos[dst * 3 + 0] - pos[src * 3 + 0];
    const float ry = pos[dst * 3 + 1] - pos[src * 3 + 1];
    const float rz = pos[dst * 3 + 2] - pos[src * 3 + 2];
    const float dist   = sqrtf(rx * rx + ry * ry + rz * rz + 1e-8f);
    const float dc     = fmaxf(dist, 1e-6f);
    const float inv_dc = 1.0f / dc;
    const float ux = rx * inv_dc, uy = ry * inv_dc, uz = rz * inv_dc;
    const float a0 = edge_attr[ec * 4 + 0], a1 = edge_attr[ec * 4 + 1];
    const float a2 = edge_attr[ec * 4 + 2], a3 = edge_attr[ec * 4 + 3];
    const float gate =
        fminf(fmaxf(1.0f + 0.6f * (leg_score(a0) * a2 + leg_score(a1) * a3), 0.35f), 2.5f);
    const float qs = charge[src], qd = charge[dst];

    auto extraf = [&](int k) -> float {
      if (k < 16)  return __sinf((float)(k + 1) * (PI_F / CUTOFF_F) * dc) * inv_dc * gate;
      if (k == 16) return dist * (1.0f / CUTOFF_F) * gate;
      if (k == 17) return qs * qd * gate;
      if (k == 18) return fabsf(qs - qd) * gate;
      return 0.0f;
    };
    v16bf aext;
#pragma unroll
    for (int i = 0; i < 8; ++i) {
      aext[i]     = (__bf16)extraf(khalf + i);
      aext[8 + i] = (__bf16)extraf(khalf + 16 + i);
    }

    // GEMM0: extras(16x32) @ Wc(32x128), B from LDS
    v8f acc[8];
#pragma unroll
    for (int nt = 0; nt < 8; ++nt) {
      const int n = nt * 16 + nlo;
      acc[nt] = wmma_bf16(aext, ldv16g(sWc + n * WC_PITCH + khalf), zero8());
    }
#pragma unroll
    for (int nt = 0; nt < 8; ++nt)
#pragma unroll
      for (int j = 0; j < 8; ++j)
        hb[(mbase + j) * 132 + nt * 16 + nlo] = acc[nt][j];
    __builtin_amdgcn_wave_barrier();

    // h = silu(extras@Wc + P[src] + Q[dst]); LN stats
    const float4* Pr = (const float4*)(P + (long)src * 128 + chalf);
    const float4* Qr = (const float4*)(Q + (long)dst * 128 + chalf);
    float s = 0.0f, s2 = 0.0f;
#pragma unroll
    for (int c4 = 0; c4 < 16; ++c4) {
      float4 pv = Pr[c4], qv = Qr[c4];
      float* hrow = &hb[nlo * 132 + chalf + c4 * 4];
      float v0 = siluf(hrow[0] + pv.x + qv.x);
      float v1 = siluf(hrow[1] + pv.y + qv.y);
      float v2 = siluf(hrow[2] + pv.z + qv.z);
      float v3 = siluf(hrow[3] + pv.w + qv.w);
      hrow[0] = v0; hrow[1] = v1; hrow[2] = v2; hrow[3] = v3;
      s  += v0 + v1 + v2 + v3;
      s2 += v0 * v0 + v1 * v1 + v2 * v2 + v3 * v3;
    }
    s  += __shfl_xor(s, 16, 32);
    s2 += __shfl_xor(s2, 16, 32);
    const float mean = s * (1.0f / 128.0f);
    const float var  = s2 * (1.0f / 128.0f) - mean * mean;
    const float rs   = rsqrtf(var + 1e-5f);
#pragma unroll
    for (int c4 = 0; c4 < 16; ++c4) {
      float* hrow = &hb[nlo * 132 + chalf + c4 * 4];
#pragma unroll
      for (int u = 0; u < 4; ++u) {
        const int c = chalf + c4 * 4 + u;
        hrow[u] = (hrow[u] - mean) * rs * g1[c] + bt1[c];
      }
    }
    __builtin_amdgcn_wave_barrier();

    // GEMM2: eh = silu(h' @ We2 + be2), B from LDS
    v8f acc2[8];
#pragma unroll
    for (int nt = 0; nt < 8; ++nt) acc2[nt] = zero8();
#pragma unroll
    for (int kc = 0; kc < 4; ++kc) {
      const int kb = kc * 32 + khalf;
      v16bf a = ldv16_lds_f32(&hb[nlo * 132], kb);
#pragma unroll
      for (int nt = 0; nt < 8; ++nt) {
        const int n = nt * 16 + nlo;
        acc2[nt] = wmma_bf16(a, ldv16g(sWe2 + n * W_PITCH + kb), acc2[nt]);
      }
    }
    __builtin_amdgcn_wave_barrier();
#pragma unroll
    for (int nt = 0; nt < 8; ++nt) {
      const int n  = nt * 16 + nlo;
      const float b2 = be2[n];
#pragma unroll
      for (int j = 0; j < 8; ++j)
        hb[(mbase + j) * 132 + n] = siluf(acc2[nt][j] + b2);
    }
    __builtin_amdgcn_wave_barrier();

    // scatter-sum eh into agg[src]; count
    if (ev) {
#pragma unroll
      for (int c4 = 0; c4 < 16; ++c4) {
        const int c = chalf + c4 * 4;
        atomicAdd(&agg[(long)src * 128 + c + 0], hb[nlo * 132 + c + 0]);
        atomicAdd(&agg[(long)src * 128 + c + 1], hb[nlo * 132 + c + 1]);
        atomicAdd(&agg[(long)src * 128 + c + 2], hb[nlo * 132 + c + 2]);
        atomicAdd(&agg[(long)src * 128 + c + 3], hb[nlo * 132 + c + 3]);
      }
      if (lane < 16) atomicAdd(&cnt[src], 1.0f);
    }

    // GEMM3: t = silu(eh @ Wc1 + bc1); coord = t @ Wc2 + bc2
    v8f acc3[4];
#pragma unroll
    for (int nt = 0; nt < 4; ++nt) acc3[nt] = zero8();
#pragma unroll
    for (int kc = 0; kc < 4; ++kc) {
      const int kb = kc * 32 + khalf;
      v16bf a = ldv16_lds_f32(&hb[nlo * 132], kb);
#pragma unroll
      for (int nt = 0; nt < 4; ++nt) {
        const int n = nt * 16 + nlo;
        acc3[nt] = wmma_bf16(a, ldv16g(sWc1 + n * W_PITCH + kb), acc3[nt]);
      }
    }
    float pj[8];
#pragma unroll
    for (int j = 0; j < 8; ++j) pj[j] = 0.0f;
#pragma unroll
    for (int nt = 0; nt < 4; ++nt) {
      const int n  = nt * 16 + nlo;
      const float w2 = Wc2[n];
      const float b  = bc1[n];
#pragma unroll
      for (int j = 0; j < 8; ++j) pj[j] += siluf(acc3[nt][j] + b) * w2;
    }
#pragma unroll
    for (int m = 1; m < 16; m <<= 1)
#pragma unroll
      for (int j = 0; j < 8; ++j) pj[j] += __shfl_xor(pj[j], m, 32);
    if (nlo == 0) {
#pragma unroll
      for (int j = 0; j < 8; ++j) cb[mbase + j] = pj[j] + bc2[0];
    }
    __builtin_amdgcn_wave_barrier();
    if (lane < 16 && ev) {
      const float cg = cb[lane] * gate;
      atomicAdd(&delta[(long)src * 3 + 0], ux * cg);
      atomicAdd(&delta[(long)src * 3 + 1], uy * cg);
      atomicAdd(&delta[(long)src * 3 + 2], uz * cg);
    }
    __builtin_amdgcn_wave_barrier();
  }
}

// ----------------------------------------------------------- node residual

__global__ __launch_bounds__(256) void k_node(
    const float* __restrict__ x, const __bf16* __restrict__ xbf,
    const float* __restrict__ agg, const float* __restrict__ cnt,
    const __bf16* __restrict__ WnaT, const __bf16* __restrict__ WnbT,
    const float* __restrict__ bn, const float* __restrict__ gn,
    const float* __restrict__ btn, float* __restrict__ xout, int N) {
  const int lane  = threadIdx.x & 31;
  const int wave  = threadIdx.x >> 5;
  const int tile  = blockIdx.x * 8 + wave;
  const int row0  = tile * 16;
  if (row0 >= N) return;
  const int nlo   = lane & 15;
  const int khalf = (lane >> 4) << 3;
  int arow = row0 + nlo; if (arow >= N) arow = N - 1;
  const float invc = 1.0f / fmaxf(cnt[arow], 1.0f);

  v8f acc[8];
#pragma unroll
  for (int nt = 0; nt < 8; ++nt) acc[nt] = zero8();
#pragma unroll
  for (int kc = 0; kc < 4; ++kc) {
    const int kb = kc * 32 + khalf;
    v16bf a1 = ldv16g(xbf + (long)arow * 128 + kb);
    v16bf a2;
#pragma unroll
    for (int i = 0; i < 8; ++i) {
      a2[i]     = (__bf16)(agg[(long)arow * 128 + kb + i] * invc);
      a2[8 + i] = (__bf16)(agg[(long)arow * 128 + kb + 16 + i] * invc);
    }
#pragma unroll
    for (int nt = 0; nt < 8; ++nt) {
      const int n = nt * 16 + nlo;
      acc[nt] = wmma_bf16(a1, ldv16g(WnaT + n * 128 + kb), acc[nt]);
      acc[nt] = wmma_bf16(a2, ldv16g(WnbT + n * 128 + kb), acc[nt]);
    }
  }
  float sj[8], s2j[8];
#pragma unroll
  for (int j = 0; j < 8; ++j) { sj[j] = 0.0f; s2j[j] = 0.0f; }
#pragma unroll
  for (int nt = 0; nt < 8; ++nt) {
    const int n   = nt * 16 + nlo;
    const float bv = bn[n];
#pragma unroll
    for (int j = 0; j < 8; ++j) {
      float v = siluf(acc[nt][j] + bv);
      acc[nt][j] = v;
      sj[j]  += v;
      s2j[j] += v * v;
    }
  }
#pragma unroll
  for (int m = 1; m < 16; m <<= 1)
#pragma unroll
    for (int j = 0; j < 8; ++j) {
      sj[j]  += __shfl_xor(sj[j], m, 32);
      s2j[j] += __shfl_xor(s2j[j], m, 32);
    }
  const int mbase = (lane < 16) ? 0 : 8;
#pragma unroll
  for (int j = 0; j < 8; ++j) {
    const int r = row0 + mbase + j;
    if (r >= N) continue;
    const float mean = sj[j] * (1.0f / 128.0f);
    const float var  = s2j[j] * (1.0f / 128.0f) - mean * mean;
    const float rs   = rsqrtf(var + 1e-5f);
#pragma unroll
    for (int nt = 0; nt < 8; ++nt) {
      const int n = nt * 16 + nlo;
      const float y = (acc[nt][j] - mean) * rs * gn[n] + btn[n];
      xout[(long)r * 128 + n] = x[(long)r * 128 + n] + y;
    }
  }
}

__global__ void k_pos(const float* __restrict__ pos, const float* __restrict__ delta,
                      const float* __restrict__ cnt, float* __restrict__ pout, int N) {
  int i = blockIdx.x * blockDim.x + threadIdx.x;
  if (i < N * 3) {
    int node = i / 3;
    float invc = 1.0f / fmaxf(cnt[node], 1.0f);
    pout[i] = pos[i] + 0.1f * delta[i] * invc;
  }
}

// ------------------------------------------------------------------- host

extern "C" void kernel_launch(void* const* d_in, const int* in_sizes, int n_in,
                              void* d_out, int out_size, void* d_ws, size_t ws_size,
                              hipStream_t stream) {
  const float* x         = (const float*)d_in[0];
  const float* pos       = (const float*)d_in[1];
  const float* charge    = (const float*)d_in[2];
  const float* edge_attr = (const float*)d_in[3];
  const int*   edge_index= (const int*)d_in[4];
  const float* We1 = (const float*)d_in[5];  const float* be1 = (const float*)d_in[6];
  const float* g1  = (const float*)d_in[7];  const float* bt1 = (const float*)d_in[8];
  const float* We2 = (const float*)d_in[9];  const float* be2 = (const float*)d_in[10];
  const float* Wn  = (const float*)d_in[11]; const float* bn  = (const float*)d_in[12];
  const float* gn  = (const float*)d_in[13]; const float* btn = (const float*)d_in[14];
  const float* Wc1 = (const float*)d_in[15]; const float* bc1 = (const float*)d_in[16];
  const float* Wc2 = (const float*)d_in[17]; const float* bc2 = (const float*)d_in[18];

  const int  N = in_sizes[0] / 128;
  const long E = (long)in_sizes[3] / 4;

  char* ws = (char*)d_ws;
  size_t off = 0;
  auto take = [&](size_t bytes) -> char* {
    char* p = ws + off;
    off = (off + bytes + 255) & ~(size_t)255;
    return p;
  };
  float*  agg   = (float*)take((size_t)N * 128 * 4);
  float*  cnt   = (float*)take((size_t)N * 4);
  float*  delta = (float*)take((size_t)N * 3 * 4);
  const long zeroFloats = (long)(off / 4);
  __bf16* xbf  = (__bf16*)take((size_t)N * 128 * 2);
  float*  P    = (float*)take((size_t)N * 128 * 4);
  float*  Q    = (float*)take((size_t)N * 128 * 4);
  __bf16* WaT  = (__bf16*)take(128 * 128 * 2);
  __bf16* WbT  = (__bf16*)take(128 * 128 * 2);
  __bf16* WcT  = (__bf16*)take(128 * 32 * 2);
  __bf16* We2T = (__bf16*)take(128 * 128 * 2);
  __bf16* Wc1T = (__bf16*)take(64 * 128 * 2);
  __bf16* WnaT = (__bf16*)take(128 * 128 * 2);
  __bf16* WnbT = (__bf16*)take(128 * 128 * 2);
  (void)ws_size; (void)n_in; (void)out_size;

  float* xout = (float*)d_out;
  float* pout = xout + (size_t)N * 128;

  {
    long g = (zeroFloats + 255) / 256;
    k_zero<<<dim3((unsigned)g), dim3(256), 0, stream>>>((float*)ws, zeroFloats);
  }
  {
    long n = (long)N * 128;
    k_cvt_bf16<<<dim3((unsigned)((n + 255) / 256)), dim3(256), 0, stream>>>(x, xbf, n);
  }
  {
    int n = 5 * 16384 + 128 * 32 + 64 * 128;
    k_prep_w<<<(n + 255) / 256, 256, 0, stream>>>(We1, We2, Wc1, Wn, WaT, WbT, WcT,
                                                  We2T, Wc1T, WnaT, WnbT);
  }
  {
    int tiles = (N + 15) / 16, blocks = (tiles + 7) / 8;
    k_nodePQ<<<blocks, 256, 0, stream>>>(xbf, WaT, WbT, be1, P, Q, N);
  }
  {
    long tiles = (E + 15) / 16;
    long blocks = (tiles + 3) / 4;
    if (blocks > 1280) blocks = 1280;   // persistent, grid-stride
    k_edge<<<dim3((unsigned)blocks), dim3(128), SM_TOTAL, stream>>>(
        pos, charge, edge_attr, edge_index, P, Q, g1, bt1, WcT, We2T, be2, Wc1T,
        bc1, Wc2, bc2, agg, cnt, delta, E);
  }
  {
    int tiles = (N + 15) / 16, blocks = (tiles + 7) / 8;
    k_node<<<blocks, 256, 0, stream>>>(x, xbf, agg, cnt, WnaT, WnbT, bn, gn, btn, xout, N);
  }
  {
    long n = (long)N * 3;
    k_pos<<<dim3((unsigned)((n + 255) / 256)), dim3(256), 0, stream>>>(pos, delta, cnt, pout, N);
  }
}
